// EdgeModule_13391708029603
// MI455X (gfx1250) — compile-verified, matrix operand order checked by reading, not run on hardware
//
#include <hip/hip_runtime.h>

typedef __attribute__((ext_vector_type(16))) __bf16        v16bf;
typedef __attribute__((ext_vector_type(2)))  __bf16        v2bf;
typedef __attribute__((ext_vector_type(8)))  float         v8f;
typedef __attribute__((ext_vector_type(8)))  unsigned int  v8u;

#define HID      128
#define K1TILES  12          // 384 / 32
#define K2TILES  4           // 128 / 32
#define NTILES   8           // 128 / 16
#define WAVES    8
#define BLOCK    (WAVES * 32)

// LDS layout (bytes), total 160 KB -> 2 workgroups per WGP (320 KB budget)
#define W1_OFF   0                       // W1 frags bf16 [12][8][32][16] = 98304 B
#define W2_OFF   98304                   // W2 frags bf16 [4][8][32][16]  = 32768 B
#define HB_OFF   131072                  // h buf per wave [8][16][128] bf16 = 32768 B
#define SMEM_BYTES 163840

#if defined(__has_builtin)
#if __has_builtin(__builtin_amdgcn_cvt_pk_bf16_f32)
#define HAVE_PK_BF16 1
#endif
#endif

// Two f32 -> packed 2 x bf16 in one 32-bit register (v_cvt_pk_bf16_f32)
__device__ __forceinline__ unsigned int cvt2u(float lo, float hi) {
#ifdef HAVE_PK_BF16
    v2bf p = __builtin_amdgcn_cvt_pk_bf16_f32(lo, hi);
    return __builtin_bit_cast(unsigned int, p);
#else
    v2bf p;
    p[0] = (__bf16)lo;
    p[1] = (__bf16)hi;
    return __builtin_bit_cast(unsigned int, p);
#endif
}

__device__ __forceinline__ unsigned short cvt1u(float f) {
    __bf16 b = (__bf16)f;
    return __builtin_bit_cast(unsigned short, b);
}

__global__ __launch_bounds__(BLOCK)
void edge_mlp_wmma(const float* __restrict__ vf,   // [N, 128]
                   const float* __restrict__ ef,   // [E, 128]
                   const int*   __restrict__ eidx, // [2, E]
                   const float* __restrict__ W1,   // [384, 128]
                   const float* __restrict__ b1,   // [128]
                   const float* __restrict__ W2,   // [128, 128]
                   const float* __restrict__ b2,   // [128]
                   float*       __restrict__ out,  // [E, 128]
                   int nEdges)
{
    extern __shared__ char smem[];
    unsigned short* w1s = (unsigned short*)(smem + W1_OFF);
    unsigned short* w2s = (unsigned short*)(smem + W2_OFF);
    unsigned short* hb  = (unsigned short*)(smem + HB_OFF);

    const int tid = threadIdx.x;

    // ---- Weight prep: swizzle W1/W2 (row-major f32) into bf16 B-fragment layout ----
    // Fragment linear layout: [kt][nt][lane][elem], 16 bf16 per lane = 32 B contiguous.
    for (int f = tid; f < K1TILES * NTILES * 32 * 16; f += BLOCK) {
        int i    = f & 15;
        int lane = (f >> 4) & 31;
        int nt   = (f >> 9) & 7;
        int kt   = f >> 12;
        int n    = nt * 16 + (lane & 15);
        int kb   = (lane >> 4) * 8;
        int k    = kt * 32 + ((i < 8) ? (kb + i) : (16 + kb + (i - 8)));
        w1s[f] = cvt1u(W1[k * HID + n]);
    }
    for (int f = tid; f < K2TILES * NTILES * 32 * 16; f += BLOCK) {
        int i    = f & 15;
        int lane = (f >> 4) & 31;
        int nt   = (f >> 9) & 7;
        int kt   = f >> 12;
        int n    = nt * 16 + (lane & 15);
        int kb   = (lane >> 4) * 8;
        int k    = kt * 32 + ((i < 8) ? (kb + i) : (16 + kb + (i - 8)));
        w2s[f] = cvt1u(W2[k * HID + n]);
    }
    __syncthreads();

    const int wave = tid >> 5;
    const int lane = tid & 31;
    const int ln15 = lane & 15;          // A: row m; B/C/D: col n
    const int kb   = (lane >> 4) * 8;    // K sub-offset for 16-bit A/B layout
    unsigned short* hw = hb + wave * (16 * HID);

    // Per-lane bias registers (C-fragment cols are n = lane&15, same for all 8 row-VGPRs)
    float bia1[NTILES], bia2[NTILES];
#pragma unroll
    for (int nt = 0; nt < NTILES; ++nt) {
        bia1[nt] = b1[nt * 16 + ln15];
        bia2[nt] = b2[nt * 16 + ln15];
    }

    const int nT = (nEdges + 15) >> 4;
    for (int tile = blockIdx.x * WAVES + wave; tile < nT; tile += gridDim.x * WAVES) {
        int e0 = tile * 16 + ln15;
        int e  = (e0 < nEdges) ? e0 : (nEdges - 1);
        int si = eidx[e];
        int ri = eidx[nEdges + e];
        const float* ps = vf + (size_t)si * HID;
        const float* pr = vf + (size_t)ri * HID;
        const float* pe = ef + (size_t)e  * HID;

        // ---- Build all 12 A fragments (16x32 bf16 each) in registers ----
        v16bf afrag[K1TILES];
#pragma unroll
        for (int kt = 0; kt < K1TILES; ++kt) {
            const float* base = (kt < 4) ? ps : (kt < 8) ? pr : pe;
            int koff = (kt & 3) * 32;
            const float4* q0 = (const float4*)(base + koff + kb);        // K = kb..kb+7
            const float4* q1 = (const float4*)(base + koff + 16 + kb);   // K = 16+kb..+7
            float4 a0 = q0[0], a1 = q0[1];
            float4 c0 = q1[0], c1 = q1[1];
            v8u u;
            u[0] = cvt2u(a0.x, a0.y); u[1] = cvt2u(a0.z, a0.w);
            u[2] = cvt2u(a1.x, a1.y); u[3] = cvt2u(a1.z, a1.w);
            u[4] = cvt2u(c0.x, c0.y); u[5] = cvt2u(c0.z, c0.w);
            u[6] = cvt2u(c1.x, c1.y); u[7] = cvt2u(c1.z, c1.w);
            afrag[kt] = __builtin_bit_cast(v16bf, u);
        }

        // ---- Layer 1: h = relu(A @ W1 + b1), h -> per-wave LDS in A layout ----
#pragma unroll
        for (int nt = 0; nt < NTILES; ++nt) {
            v8f acc;
#pragma unroll
            for (int j = 0; j < 8; ++j) acc[j] = bia1[nt];
#pragma unroll
            for (int kt = 0; kt < K1TILES; ++kt) {
                const v8u* bp = (const v8u*)(smem + W1_OFF + (((kt * 8 + nt) * 32 + lane) << 5));
                v16bf bfrag = __builtin_bit_cast(v16bf, *bp);
                acc = __builtin_amdgcn_wmma_f32_16x16x32_bf16(
                          false, afrag[kt], false, bfrag, (short)0, acc, false, false);
            }
            int colbase = nt * 16 + ln15;  // D layout: lane carries column n
#pragma unroll
            for (int j = 0; j < 8; j += 2) {
                float v0 = fmaxf(acc[j],     0.f);       // ReLU
                float v1 = fmaxf(acc[j + 1], 0.f);
                unsigned int p = cvt2u(v0, v1);          // one packed cvt per row pair
                int row = j + 8 * (lane >> 4);           // D layout: VGPR j -> row
                hw[row * HID + colbase]       = (unsigned short)p;
                hw[(row + 1) * HID + colbase] = (unsigned short)(p >> 16);
            }
        }

        // ---- Layer 2 A fragments from h buffer (wave-private LDS, DS in-order) ----
        v16bf a2[K2TILES];
#pragma unroll
        for (int kt = 0; kt < K2TILES; ++kt) {
            uint4 lo = *(const uint4*)(hw + ln15 * HID + kt * 32 + kb);
            uint4 hi = *(const uint4*)(hw + ln15 * HID + kt * 32 + 16 + kb);
            v8u u;
            u[0] = lo.x; u[1] = lo.y; u[2] = lo.z; u[3] = lo.w;
            u[4] = hi.x; u[5] = hi.y; u[6] = hi.z; u[7] = hi.w;
            a2[kt] = __builtin_bit_cast(v16bf, u);
        }

        // ---- Layer 2: out = h @ W2 + b2 ----
#pragma unroll
        for (int nt = 0; nt < NTILES; ++nt) {
            v8f acc;
#pragma unroll
            for (int j = 0; j < 8; ++j) acc[j] = bia2[nt];
#pragma unroll
            for (int kt = 0; kt < K2TILES; ++kt) {
                const v8u* bp = (const v8u*)(smem + W2_OFF + (((kt * 8 + nt) * 32 + lane) << 5));
                v16bf bfrag = __builtin_bit_cast(v16bf, *bp);
                acc = __builtin_amdgcn_wmma_f32_16x16x32_bf16(
                          false, a2[kt], false, bfrag, (short)0, acc, false, false);
            }
#pragma unroll
            for (int j = 0; j < 8; ++j) {
                int row = j + 8 * (lane >> 4);
                int ee  = tile * 16 + row;
                if (ee < nEdges)
                    out[(size_t)ee * HID + nt * 16 + ln15] = acc[j];
            }
        }
    }
}

extern "C" void kernel_launch(void* const* d_in, const int* in_sizes, int n_in,
                              void* d_out, int out_size, void* d_ws, size_t ws_size,
                              hipStream_t stream) {
    (void)n_in; (void)out_size; (void)d_ws; (void)ws_size;
    const float* vf   = (const float*)d_in[0];
    const float* ef   = (const float*)d_in[1];
    const int*   eidx = (const int*)  d_in[2];
    const float* W1   = (const float*)d_in[3];
    const float* b1   = (const float*)d_in[4];
    const float* W2   = (const float*)d_in[5];
    const float* b2   = (const float*)d_in[6];
    float*       out  = (float*)d_out;

    int nEdges = in_sizes[1] / HID;
    int nT     = (nEdges + 15) / 16;
    int blocks = (nT + WAVES - 1) / WAVES;
    if (blocks > 1024) blocks = 1024;   // persistent grid; weight prep amortized
    if (blocks < 1)    blocks = 1;

    hipFuncSetAttribute(reinterpret_cast<const void*>(&edge_mlp_wmma),
                        hipFuncAttributeMaxDynamicSharedMemorySize, SMEM_BYTES);

    edge_mlp_wmma<<<blocks, BLOCK, SMEM_BYTES, stream>>>(
        vf, ef, eidx, W1, b1, W2, b2, out, nEdges);
}
